// SimpleInteractionBlock1_21019569947168
// MI455X (gfx1250) — matrix operand order repaired and checked by reading, not compiled, not run on hardware
//
#include <hip/hip_runtime.h>
#include <hip/hip_bf16.h>

typedef __bf16 bf16;
typedef __attribute__((ext_vector_type(16))) __bf16 v16bf;
typedef __attribute__((ext_vector_type(8)))  float  v8f;

#define LDSB 136   // LDS row stride in bf16 elements (128 + 8 pad); 272B = 16B aligned

__device__ __forceinline__ float swishf(float v) {
    return v / (1.0f + __expf(-v));
}

// One wave: 16xK @ Kx128 -> 16x128 (8 accumulators of 16x16), bf16 in, f32 acc.
// Fragment layouts per CDNA5 ISA 7.12.2 (wave32).
__device__ __forceinline__ void wave_gemm(const bf16* __restrict__ A, int lda,
                                          const bf16* __restrict__ B, int ldb,
                                          int K, v8f* acc) {
    const int lane = threadIdx.x & 31;
    const int m    = lane & 15;
    const int hf   = lane >> 4;
    for (int k0 = 0; k0 < K; k0 += 32) {
        v16bf a;
        const bf16* arow = A + m * lda + k0 + 8 * hf;
#pragma unroll
        for (int g = 0; g < 8; ++g) {
            int kk = (g < 4) ? (2 * g) : (16 + 2 * (g - 4));
            a[2 * g]     = arow[kk];
            a[2 * g + 1] = arow[kk + 1];
        }
        const bf16* brow = B + (size_t)(k0 + 8 * hf) * ldb + m;
#pragma unroll
        for (int nt = 0; nt < 8; ++nt) {
            v16bf b;
#pragma unroll
            for (int e = 0; e < 16; ++e) b[e] = brow[(size_t)((e < 8) ? e : (e + 8)) * ldb + nt * 16];
            acc[nt] = __builtin_amdgcn_wmma_f32_16x16x32_bf16(
                false, a, false, b, (short)0, acc[nt], false, false);
        }
    }
}

__device__ __forceinline__ void zero_acc(v8f* acc) {
#pragma unroll
    for (int i = 0; i < 8; ++i)
#pragma unroll
        for (int j = 0; j < 8; ++j) acc[i][j] = 0.0f;
}

// Stage W^T (W is [128,K] row-major, row stride ldw) into LDS bf16 [K][LDSB].
// Coalesced float4 reads along k, transposed scatter into LDS.
template<int K>
__device__ __forceinline__ void stage_w(const float* __restrict__ W, int ldw,
                                        bf16* __restrict__ Wt, int tid, int nthr) {
    constexpr int KV = K / 4;
    for (int i = tid; i < 128 * KV; i += nthr) {
        int o = i / KV, k = (i - o * KV) * 4;
        const float4 w4 = *(const float4*)(W + (size_t)o * ldw + k);
        Wt[(k + 0) * LDSB + o] = (bf16)w4.x;
        Wt[(k + 1) * LDSB + o] = (bf16)w4.y;
        Wt[(k + 2) * LDSB + o] = (bf16)w4.z;
        Wt[(k + 3) * LDSB + o] = (bf16)w4.w;
    }
}

// Epilogue: out = act(acc + bias + addA) + addB, with optional row-bounds check.
template<bool CHECK, bool ACT, bool BIAS, bool ADDA, bool ADDB>
__device__ __forceinline__ void store_tile(const v8f* acc, float* __restrict__ out,
                                           const float* __restrict__ bias,
                                           const float* __restrict__ addA,
                                           const float* __restrict__ addB,
                                           int rowbase, int nrows, int lane) {
    const int hf = lane >> 4, n15 = lane & 15;
#pragma unroll
    for (int nt = 0; nt < 8; ++nt) {
        const int col = nt * 16 + n15;
        const float bv = BIAS ? bias[col] : 0.0f;
#pragma unroll
        for (int g = 0; g < 8; ++g) {
            const int row = rowbase + g + 8 * hf;
            if (!CHECK || row < nrows) {
                const size_t idx = (size_t)row * 128 + col;
                float v = acc[nt][g] + bv;
                if (ADDA) v += addA[idx];
                if (ACT)  v = swishf(v);
                if (ADDB) v += addB[idx];
                out[idx] = v;
            }
        }
    }
}

// Single-input GEMM: out = act(X @ W^T + bias) (+addB), optional per-col
// affine on X at staging (GraphNorm folding). 256 thr / 8 waves / 128 rows.
template<int K, bool ACT, bool BIAS, bool ADDB, bool SCALE>
__global__ __launch_bounds__(256)
void node_gemm_k(const float* __restrict__ X, int ldx,
                 const float* __restrict__ W, int ldw,
                 const float* __restrict__ bias,
                 const float* __restrict__ addB,
                 const float* __restrict__ colscale,
                 const float* __restrict__ colshift,
                 float* __restrict__ out, int nrows) {
    __shared__ bf16 Wt[K * LDSB];
    __shared__ bf16 At[8][16 * LDSB];
    const int tid = threadIdx.x, wave = tid >> 5, lane = tid & 31;

    stage_w<K>(W, ldw, Wt, tid, 256);

    const int rowbase = blockIdx.x * 128 + wave * 16;
    constexpr int KV = K / 4;
    for (int i = lane; i < 16 * KV; i += 32) {
        int r = i / KV, k = (i - r * KV) * 4;
        int row = rowbase + r;
        float4 v = (row < nrows) ? *(const float4*)(X + (size_t)row * ldx + k)
                                 : make_float4(0.f, 0.f, 0.f, 0.f);
        if (SCALE) {
            v.x = v.x * colscale[k + 0] + colshift[k + 0];
            v.y = v.y * colscale[k + 1] + colshift[k + 1];
            v.z = v.z * colscale[k + 2] + colshift[k + 2];
            v.w = v.w * colscale[k + 3] + colshift[k + 3];
        }
        At[wave][r * LDSB + k + 0] = (bf16)v.x;
        At[wave][r * LDSB + k + 1] = (bf16)v.y;
        At[wave][r * LDSB + k + 2] = (bf16)v.z;
        At[wave][r * LDSB + k + 3] = (bf16)v.w;
    }
    __syncthreads();

    v8f acc[8];
    zero_acc(acc);
    wave_gemm(At[wave], LDSB, Wt, LDSB, K, acc);

    if (rowbase + 16 <= nrows)
        store_tile<false, ACT, BIAS, false, ADDB>(acc, out, bias, nullptr, addB,
                                                  rowbase, nrows, lane);
    else
        store_tile<true, ACT, BIAS, false, ADDB>(acc, out, bias, nullptr, addB,
                                                 rowbase, nrows, lane);
}

// Dual-input GEMM: out = X1 @ W1^T + X2 @ W2^T + bias (+addA).
// Fuses conv_rel+conv_root and the two halves of lincat (kills T round-trips).
template<bool ADDA>
__global__ __launch_bounds__(256)
void node_dual_k(const float* __restrict__ X1, const float* __restrict__ W1, int ldw1,
                 const float* __restrict__ X2, const float* __restrict__ W2, int ldw2,
                 const float* __restrict__ bias, const float* __restrict__ addA,
                 float* __restrict__ out, int nrows) {
    __shared__ bf16 W1t[128 * LDSB];
    __shared__ bf16 W2t[128 * LDSB];
    __shared__ bf16 A1t[8][16 * LDSB];
    __shared__ bf16 A2t[8][16 * LDSB];
    const int tid = threadIdx.x, wave = tid >> 5, lane = tid & 31;

    stage_w<128>(W1, ldw1, W1t, tid, 256);
    stage_w<128>(W2, ldw2, W2t, tid, 256);

    const int rowbase = blockIdx.x * 128 + wave * 16;
    for (int i = lane; i < 16 * 32; i += 32) {
        int r = i / 32, k = (i - r * 32) * 4;
        int row = rowbase + r;
        float4 v1 = (row < nrows) ? *(const float4*)(X1 + (size_t)row * 128 + k)
                                  : make_float4(0.f, 0.f, 0.f, 0.f);
        float4 v2 = (row < nrows) ? *(const float4*)(X2 + (size_t)row * 128 + k)
                                  : make_float4(0.f, 0.f, 0.f, 0.f);
        A1t[wave][r * LDSB + k + 0] = (bf16)v1.x;
        A1t[wave][r * LDSB + k + 1] = (bf16)v1.y;
        A1t[wave][r * LDSB + k + 2] = (bf16)v1.z;
        A1t[wave][r * LDSB + k + 3] = (bf16)v1.w;
        A2t[wave][r * LDSB + k + 0] = (bf16)v2.x;
        A2t[wave][r * LDSB + k + 1] = (bf16)v2.y;
        A2t[wave][r * LDSB + k + 2] = (bf16)v2.z;
        A2t[wave][r * LDSB + k + 3] = (bf16)v2.w;
    }
    __syncthreads();

    v8f acc[8];
    zero_acc(acc);
    wave_gemm(A1t[wave], LDSB, W1t, LDSB, 128, acc);
    wave_gemm(A2t[wave], LDSB, W2t, LDSB, 128, acc);

    if (rowbase + 16 <= nrows)
        store_tile<false, false, true, ADDA, false>(acc, out, bias, addA, nullptr,
                                                    rowbase, nrows, lane);
    else
        store_tile<true, false, true, ADDA, false>(acc, out, bias, addA, nullptr,
                                                   rowbase, nrows, lane);
}

// Fused edge pipeline: f = (feat @ W1^T) @ W2^T per 16-edge wave tile fully
// on-chip, then agg[dst] += f * xp[src] via hardware FP32 atomics.
template<int F>
__global__ __launch_bounds__(256)
void edge_scatter_k(const float* __restrict__ feat,
                    const float* __restrict__ W1,
                    const float* __restrict__ W2,
                    const int* __restrict__ src, const int* __restrict__ dst,
                    const float* __restrict__ xp,
                    float* __restrict__ agg, int E) {
    __shared__ bf16 W1t[F * LDSB];
    __shared__ bf16 W2t[128 * LDSB];
    __shared__ bf16 Ft[8][16 * LDSB];
    __shared__ bf16 Tt[8][16 * LDSB];
    const int tid = threadIdx.x, wave = tid >> 5, lane = tid & 31;

    stage_w<F>(W1, F, W1t, tid, 256);
    stage_w<128>(W2, 128, W2t, tid, 256);

    const int ebase = blockIdx.x * 128 + wave * 16;
    constexpr int FV = F / 4;
    for (int i = lane; i < 16 * FV; i += 32) {
        int r = i / FV, k = (i - r * FV) * 4;
        int e = ebase + r;
        float4 v = (e < E) ? *(const float4*)(feat + (size_t)e * F + k)
                           : make_float4(0.f, 0.f, 0.f, 0.f);
        Ft[wave][r * LDSB + k + 0] = (bf16)v.x;
        Ft[wave][r * LDSB + k + 1] = (bf16)v.y;
        Ft[wave][r * LDSB + k + 2] = (bf16)v.z;
        Ft[wave][r * LDSB + k + 3] = (bf16)v.w;
    }
    __syncthreads();

    v8f acc[8];
    zero_acc(acc);
    wave_gemm(Ft[wave], LDSB, W1t, LDSB, F, acc);       // t = feat @ W1^T

    const int hf = lane >> 4, n15 = lane & 15;
    // Spill t to per-wave LDS tile as bf16 (same-wave DS ops are in-order).
#pragma unroll
    for (int nt = 0; nt < 8; ++nt)
#pragma unroll
        for (int g = 0; g < 8; ++g)
            Tt[wave][(g + 8 * hf) * LDSB + nt * 16 + n15] = (bf16)acc[nt][g];

    zero_acc(acc);
    wave_gemm(Tt[wave], LDSB, W2t, LDSB, 128, acc);     // f = t @ W2^T

    int s8[8], d8[8];
#pragma unroll
    for (int g = 0; g < 8; ++g) {
        int e = ebase + g + 8 * hf;
        s8[g] = (e < E) ? src[e] : -1;
        d8[g] = (e < E) ? dst[e] : -1;
    }
#pragma unroll
    for (int nt = 0; nt < 8; ++nt) {
#pragma unroll
        for (int g = 0; g < 8; ++g) {
            if (d8[g] >= 0) {
                int col = nt * 16 + n15;
                float v = acc[nt][g] * xp[(size_t)s8[g] * 128 + col];
                unsafeAtomicAdd(&agg[(size_t)d8[g] * 128 + col], v);
            }
        }
    }
}

// Per-column sum / sum-of-squares over nrows (GraphNorm statistics).
__global__ __launch_bounds__(256)
void colstats_k(const float* __restrict__ h, int nrows, float* __restrict__ stats) {
    __shared__ float red[512];
    const int tid = threadIdx.x;
    const int col = tid & 127, ph = tid >> 7;
    float s = 0.0f, s2 = 0.0f;
    for (int row = blockIdx.x * 2 + ph; row < nrows; row += gridDim.x * 2) {
        float v = h[(size_t)row * 128 + col];
        s += v; s2 += v * v;
    }
    red[tid] = s;
    red[256 + tid] = s2;
    __syncthreads();
    if (ph == 0) {
        s  = red[col] + red[128 + col];
        s2 = red[256 + col] + red[256 + 128 + col];
        unsafeAtomicAdd(&stats[col], s);
        unsafeAtomicAdd(&stats[128 + col], s2);
    }
}

// Fold GraphNorm into per-column affine: norm(h) = alpha*h + beta
__global__ void affine_k(const float* __restrict__ stats,
                         const float* __restrict__ w, const float* __restrict__ b,
                         const float* __restrict__ ms, float* __restrict__ aff,
                         int nrows) {
    int c = threadIdx.x;
    if (c < 128) {
        float inv_n = 1.0f / (float)nrows;
        float mean  = stats[c] * inv_n;
        float ex2   = stats[128 + c] * inv_n;
        float mms   = mean * ms[c];
        float var   = ex2 - 2.0f * mms * mean + mms * mms;
        float alpha = w[c] * rsqrtf(var + 1e-5f);
        aff[c]       = alpha;
        aff[128 + c] = b[c] - alpha * mms;
    }
}

__global__ void fill_k(float* __restrict__ p, size_t n) {
    size_t i = (size_t)blockIdx.x * blockDim.x + threadIdx.x;
    size_t st = (size_t)gridDim.x * blockDim.x;
    for (; i < n; i += st) p[i] = 0.0f;
}

extern "C" void kernel_launch(void* const* d_in, const int* in_sizes, int n_in,
                              void* d_out, int out_size, void* d_ws, size_t ws_size,
                              hipStream_t stream) {
    const float* x            = (const float*)d_in[0];
    const float* feature1     = (const float*)d_in[1];
    const float* feature2     = (const float*)d_in[2];
    const int*   edge_index   = (const int*)d_in[3];
    const float* lin_w        = (const float*)d_in[4];
    const float* lin_b        = (const float*)d_in[5];
    const float* f1_w1        = (const float*)d_in[6];
    const float* f1_w2        = (const float*)d_in[7];
    const float* f2_w1        = (const float*)d_in[8];
    const float* f2_w2        = (const float*)d_in[9];
    const float* conv1_rel_w  = (const float*)d_in[10];
    const float* conv1_rel_b  = (const float*)d_in[11];
    const float* conv1_root_w = (const float*)d_in[12];
    const float* conv2_rel_w  = (const float*)d_in[13];
    const float* conv2_rel_b  = (const float*)d_in[14];
    const float* conv2_root_w = (const float*)d_in[15];
    const float* lin1_w       = (const float*)d_in[16];
    const float* lin1_b       = (const float*)d_in[17];
    const float* lin2_w       = (const float*)d_in[18];
    const float* lin2_b       = (const float*)d_in[19];
    const float* lincat_w     = (const float*)d_in[20];
    const float* lincat_b     = (const float*)d_in[21];
    const float* lins_w       = (const float*)d_in[22];
    const float* lins_b       = (const float*)d_in[23];
    const float* norm_weight  = (const float*)d_in[24];
    const float* norm_bias    = (const float*)d_in[25];
    const float* norm_mscale  = (const float*)d_in[26];
    const float* final_w      = (const float*)d_in[27];
    const float* final_b      = (const float*)d_in[28];

    const int N = in_sizes[0] / 128;
    const int E = in_sizes[1] / 128;   // F1 == 128
    const int* src = edge_index;
    const int* dst = edge_index + E;

    const size_t N128 = (size_t)N * 128;
    float* XP    = (float*)d_ws;
    float* AGG1  = XP + N128;
    float* AGG2  = AGG1 + N128;
    float* T     = AGG2 + N128;
    float* H1    = T + N128;
    float* H2    = H1 + N128;
    float* stats = H2 + N128;     // 256 floats
    float* aff   = stats + 256;   // 256 floats
    float* out   = (float*)d_out;

    const int nodeBlocks = (N + 127) / 128;
    const int edgeBlocks = (E + 127) / 128;
    const float* NUL = nullptr;

    // x' = swish(x @ lin_w^T + lin_b)
    node_gemm_k<128, true, true, false, false><<<nodeBlocks, 256, 0, stream>>>(
        x, 128, lin_w, 128, lin_b, NUL, NUL, NUL, XP, N);
    // zero accumulators / stats
    fill_k<<<2048, 256, 0, stream>>>(AGG1, N128);
    fill_k<<<2048, 256, 0, stream>>>(AGG2, N128);
    fill_k<<<1, 256, 0, stream>>>(stats, 256);

    // fused edge GEMM-chain + gather/scatter
    edge_scatter_k<128><<<edgeBlocks, 256, 0, stream>>>(feature1, f1_w1, f1_w2,
                                                        src, dst, XP, AGG1, E);
    edge_scatter_k<32><<<edgeBlocks, 256, 0, stream>>>(feature2, f2_w1, f2_w2,
                                                       src, dst, XP, AGG2, E);

    // conv1 fused: T = agg1@rel1^T + x'@root1^T + b1 ; h1 = swish(T@lin1^T + lin1_b)
    node_dual_k<false><<<nodeBlocks, 256, 0, stream>>>(
        AGG1, conv1_rel_w, 128, XP, conv1_root_w, 128, conv1_rel_b, NUL, T, N);
    node_gemm_k<128, true, true, false, false><<<nodeBlocks, 256, 0, stream>>>(
        T, 128, lin1_w, 128, lin1_b, NUL, NUL, NUL, H1, N);
    // conv2 fused
    node_dual_k<false><<<nodeBlocks, 256, 0, stream>>>(
        AGG2, conv2_rel_w, 128, XP, conv2_root_w, 128, conv2_rel_b, NUL, T, N);
    node_gemm_k<128, true, true, false, false><<<nodeBlocks, 256, 0, stream>>>(
        T, 128, lin2_w, 128, lin2_b, NUL, NUL, NUL, H2, N);
    // h = h1@lincat[:, :128]^T + h2@lincat[:,128:]^T + lincat_b + x'  (fused)
    node_dual_k<true><<<nodeBlocks, 256, 0, stream>>>(
        H1, lincat_w, 256, H2, lincat_w + 128, 256, lincat_b, XP, T, N);
    // 3 residual swish layers (ping-pong T <-> H1)
    node_gemm_k<128, true, true, true, false><<<nodeBlocks, 256, 0, stream>>>(
        T, 128, lins_w, 128, lins_b, T, NUL, NUL, H1, N);
    node_gemm_k<128, true, true, true, false><<<nodeBlocks, 256, 0, stream>>>(
        H1, 128, lins_w + 128 * 128, 128, lins_b + 128, H1, NUL, NUL, T, N);
    node_gemm_k<128, true, true, true, false><<<nodeBlocks, 256, 0, stream>>>(
        T, 128, lins_w + 2 * 128 * 128, 128, lins_b + 256, T, NUL, NUL, H1, N);
    // GraphNorm -> per-column affine, folded into final GEMM's A-staging
    colstats_k<<<512, 256, 0, stream>>>(H1, N, stats);
    affine_k<<<1, 128, 0, stream>>>(stats, norm_weight, norm_bias, norm_mscale, aff, N);
    node_gemm_k<128, false, true, false, true><<<nodeBlocks, 256, 0, stream>>>(
        H1, 128, final_w, 128, final_b, NUL, aff, aff + 128, out, N);
}
